// SpatioTemporalNet_87857851007248
// MI455X (gfx1250) — matrix operand order repaired and checked by reading, not compiled, not run on hardware
//
#include <hip/hip_runtime.h>
#include <math.h>

// ---------------- dims ----------------
#define D1 60
#define D2 15
#define D3 5
#define NVOX 4500            // D1*D2*D3
#define FEAT 9000            // 2*NVOX
#define HID 512
#define GATES 2048           // 4*HID
#define BATCH 4
#define TIN 5
#define TOUT 3
#define MROWS 20             // TIN*BATCH
#define NSORT 8192

typedef __attribute__((ext_vector_type(16))) _Float16 v16h;
typedef __attribute__((ext_vector_type(8)))  float    v8f;

// Wave-internal LDS fence: stores by lanes of this wave visible to its own gathers.
__device__ inline void wave_lds_fence() {
    asm volatile("s_wait_dscnt 0" ::: "memory");
}

// Load 16 consecutive f32 (4x float4) and pack to 16 f16.
__device__ inline v16h pack16(const float* src) {
    const float4* s4 = (const float4*)src;
    v16h r;
#pragma unroll
    for (int i = 0; i < 4; ++i) {
        const float4 v = s4[i];
        r[4 * i + 0] = (_Float16)v.x;
        r[4 * i + 1] = (_Float16)v.y;
        r[4 * i + 2] = (_Float16)v.z;
        r[4 * i + 3] = (_Float16)v.w;
    }
    return r;
}

// Build A fragment (16x32 f16, M x K) from f16 LDS tile a[m*ld + k], ISA 7.12.2:
// lanes 0-15: M=lane, K in {0..7,16..23}; lanes 16-31: M=lane-16, K in {8..15,24..31}
__device__ inline v16h frag_A(const _Float16* a, int ld) {
    const int lane = threadIdx.x & 31;
    const int grp  = lane >> 4;
    const int m    = lane & 15;
    v16h r;
#pragma unroll
    for (int h = 0; h < 16; ++h) {
        const int v = h >> 1, p = h & 1;
        const int k = ((v & 4) << 2) + (grp << 3) + ((v & 3) << 1) + p;
        r[h] = a[m * ld + k];
    }
    return r;
}

// Build B fragment (32x16 f16, K x N) from f16 LDS tile stored n-major: b[n*ld + k].
// lanes 0-15: N=lane, K in {0..7,16..23} ; lanes 16-31: N=lane-16, K in {8..15,24..31}
__device__ inline v16h frag_B(const _Float16* b, int ld) {
    const int lane = threadIdx.x & 31;
    const int grp  = lane >> 4;
    const int n    = lane & 15;
    v16h r;
#pragma unroll
    for (int h = 0; h < 16; ++h) {
        const int k = ((h & 8) << 1) + (grp << 3) + (h & 7);
        r[h] = b[n * ld + k];
    }
    return r;
}

// ---------------------------------------------------------------------------
// Kernel 1: G0[m=t*4+b][j] = x_flat[b,t,:] @ Wih0^T + bih0 + bhh0   (20 x 2048)
// grid 128 (16-col tiles), 256 thr (8 waves split K=9000)
// ---------------------------------------------------------------------------
__global__ __launch_bounds__(256) void k_gemm_g0(
    const float* __restrict__ x, const float* __restrict__ Wih0,
    const float* __restrict__ bih0, const float* __restrict__ bhh0,
    float* __restrict__ G0)
{
    __shared__ __attribute__((aligned(32))) _Float16 smA[8][32 * 32];
    __shared__ __attribute__((aligned(32))) _Float16 smB[8][16 * 32];
    __shared__ float smR[8][512];
    const int tid = threadIdx.x;
    const int w = tid >> 5, lane = tid & 31;
    const int nblk = blockIdx.x;
    _Float16* A  = smA[w];
    _Float16* Bt = smB[w];
    v8f c_lo = {}; v8f c_hi = {};

    // pre-zero pad rows of A (rows >= MROWS never rewritten)
    if (lane >= MROWS) {
        const v16h z = {};
        *(v16h*)(A + lane * 32)      = z;
        *(v16h*)(A + lane * 32 + 16) = z;
    }

    const int nch = (FEAT + 31) / 32;           // 282 (last one partial: 8 valid)
    for (int ch = w; ch < nch; ch += 8) {
        const int k0 = ch * 32;
        if (k0 + 32 <= FEAT) {
            // fast path: no guards, float4 loads, packed f16 stores
            if (lane < MROWS) {
                const int bb = lane & 3, tt = lane >> 2;
                const float* src = x + ((size_t)(bb * TIN + tt)) * FEAT + k0;
                *(v16h*)(A + lane * 32)      = pack16(src);
                *(v16h*)(A + lane * 32 + 16) = pack16(src + 16);
            }
            {
                const int n = lane & 15, hh = lane >> 4;
                const int j = nblk * 16 + n;
                const float* src = Wih0 + (size_t)j * FEAT + k0 + hh * 16;
                *(v16h*)(Bt + n * 32 + hh * 16) = pack16(src);
            }
        } else {
            // tail chunk (runs once per wave at most)
            if (lane < MROWS) {
                const int bb = lane & 3, tt = lane >> 2;
                const float* src = x + ((size_t)(bb * TIN + tt)) * FEAT + k0;
#pragma unroll
                for (int q = 0; q < 32; ++q)
                    A[lane * 32 + q] = (k0 + q < FEAT) ? (_Float16)src[q] : (_Float16)0.f;
            }
            {
                const int n = lane & 15, hh = lane >> 4;
                const int j = nblk * 16 + n;
                const float* src = Wih0 + (size_t)j * FEAT + k0 + hh * 16;
#pragma unroll
                for (int q = 0; q < 16; ++q) {
                    const int k = k0 + hh * 16 + q;
                    Bt[n * 32 + hh * 16 + q] = (k < FEAT) ? (_Float16)src[q] : (_Float16)0.f;
                }
            }
        }
        wave_lds_fence();
        v16h bf = frag_B(Bt, 32);
        v16h a0 = frag_A(A, 32);
        v16h a1 = frag_A(A + 16 * 32, 32);
        c_lo = __builtin_amdgcn_wmma_f32_16x16x32_f16(false, a0, false, bf, (short)0, c_lo, false, false);
        c_hi = __builtin_amdgcn_wmma_f32_16x16x32_f16(false, a1, false, bf, (short)0, c_hi, false, false);
    }
    // per-wave partials -> LDS, reduce across 8 waves
    {
        const int grp = lane >> 4, n = lane & 15;
#pragma unroll
        for (int r = 0; r < 8; ++r) {
            const int m = grp * 8 + r;
            smR[w][m * 16 + n] = c_lo[r];
        }
    }
    __syncthreads();
    {   // reduce tile 0 (rows 0..15)
        float acc = 0.f;
#pragma unroll
        for (int wv = 0; wv < 8; ++wv) acc += smR[wv][tid & 255];
        const int m = tid >> 4;          // 0..15
        const int j = nblk * 16 + (tid & 15);
        if (m < 16) G0[m * GATES + j] = acc + bih0[j] + bhh0[j];
    }
    __syncthreads();
    {   // swap in tile-1 partials and reduce rows 16..19
        const int grp = lane >> 4, n = lane & 15;
#pragma unroll
        for (int r = 0; r < 8; ++r) {
            const int m = grp * 8 + r;
            smR[w][m * 16 + n] = c_hi[r];
        }
    }
    __syncthreads();
    {
        float acc = 0.f;
#pragma unroll
        for (int wv = 0; wv < 8; ++wv) acc += smR[wv][tid & 255];
        const int m = 16 + (tid >> 4);
        const int j = nblk * 16 + (tid & 15);
        if (m < MROWS) G0[m * GATES + j] = acc + bih0[j] + bhh0[j];
    }
}

// ---------------------------------------------------------------------------
// Kernel 2: one LSTM step. gates = [xside@Wx^T] + [hprev@Whh^T] + (Gpre | ba+bb)
// grid 32 (16 hidden units each), 128 thr = 4 waves (one per gate i,f,g,o)
// ---------------------------------------------------------------------------
__global__ __launch_bounds__(128) void k_lstm_step(
    const float* __restrict__ xside, const float* __restrict__ Wx,
    const float* __restrict__ Gpre,
    const float* __restrict__ hprev, const float* __restrict__ Whh,
    const float* __restrict__ ba, const float* __restrict__ bb,
    float* __restrict__ cstate, float* __restrict__ hout, int first)
{
    __shared__ __attribute__((aligned(32))) _Float16 smA[4][16 * 32];
    __shared__ __attribute__((aligned(32))) _Float16 smB[4][16 * 32];
    __shared__ float gbuf[4][256];
    const int tid = threadIdx.x, w = tid >> 5, lane = tid & 31;
    const int u0 = blockIdx.x * 16;
    const int gate = w;
    _Float16* A  = smA[w];
    _Float16* Bt = smB[w];
    v8f c = {};

    // pre-zero A rows 4..15 (never rewritten)
    if (lane >= BATCH && lane < 16) {
        const v16h z = {};
        *(v16h*)(A + lane * 32)      = z;
        *(v16h*)(A + lane * 32 + 16) = z;
    }

#pragma unroll
    for (int pass = 0; pass < 2; ++pass) {
        const float* av = pass ? hprev : xside;
        const float* bw = pass ? Whh   : Wx;
        if (av == nullptr || bw == nullptr) continue;
        for (int ch = 0; ch < 16; ++ch) {
            const int k0 = ch * 32;
            if (lane < 8) {       // A rows 0..3, 16 floats per lane
                const int m = lane >> 1, hh = lane & 1;
                const float* src = av + m * HID + k0 + hh * 16;
                *(v16h*)(A + m * 32 + hh * 16) = pack16(src);
            }
            {                     // B^T rows n, weight row j = gate*512 + u0 + n
                const int n = lane & 15, hh = lane >> 4;
                const int j = gate * HID + u0 + n;
                const float* src = bw + (size_t)j * HID + k0 + hh * 16;
                *(v16h*)(Bt + n * 32 + hh * 16) = pack16(src);
            }
            wave_lds_fence();
            v16h af = frag_A(A, 32);
            v16h bf = frag_B(Bt, 32);
            c = __builtin_amdgcn_wmma_f32_16x16x32_f16(false, af, false, bf, (short)0, c, false, false);
        }
    }
    {
        const int grp = lane >> 4, n = lane & 15;
#pragma unroll
        for (int r = 0; r < 8; ++r) {
            const int m = grp * 8 + r;
            if (m < BATCH) gbuf[w][m * 16 + n] = c[r];
        }
    }
    __syncthreads();
    if (tid < 64) {
        const int b = tid >> 4, n = tid & 15, u = u0 + n;
        float gv[4];
#pragma unroll
        for (int g = 0; g < 4; ++g) {
            float v = gbuf[g][b * 16 + n];
            v += (Gpre != nullptr) ? Gpre[b * GATES + g * HID + u]
                                   : (ba[g * HID + u] + bb[g * HID + u]);
            gv[g] = v;
        }
        const float ig = 1.f / (1.f + __expf(-gv[0]));
        const float fg = 1.f / (1.f + __expf(-gv[1]));
        const float gg = tanhf(gv[2]);
        const float og = 1.f / (1.f + __expf(-gv[3]));
        const float cold = first ? 0.f : cstate[b * HID + u];
        const float cnew = fg * cold + ig * gg;
        cstate[b * HID + u] = cnew;
        hout[b * HID + u]   = og * tanhf(cnew);
    }
}

// ---------------------------------------------------------------------------
// Kernel 3: attention scores e[b,t] = A2w . tanh(A1w @ h + A1b) + A2b
// ---------------------------------------------------------------------------
__global__ __launch_bounds__(256) void k_att_scores(
    const float* __restrict__ h2s, const float* __restrict__ A1w,
    const float* __restrict__ A1b, const float* __restrict__ A2w,
    const float* __restrict__ A2b, float* __restrict__ esc)
{
    __shared__ float hsh[HID];
    __shared__ float red[256];
    const int blk = blockIdx.x;            // b*5 + t
    const int b = blk / TIN, t = blk % TIN;
    const int tid = threadIdx.x;
    const float* h = h2s + t * (BATCH * HID) + b * HID;
    for (int j = tid; j < HID; j += 256) hsh[j] = h[j];
    __syncthreads();
    float part = 0.f;
    for (int j = tid; j < HID; j += 256) {
        const float* wr = A1w + (size_t)j * HID;
        float acc = A1b[j];
        for (int k = 0; k < HID; ++k) acc += wr[k] * hsh[k];
        part += A2w[j] * tanhf(acc);
    }
    red[tid] = part;
    __syncthreads();
    for (int s = 128; s > 0; s >>= 1) {
        if (tid < s) red[tid] += red[tid + s];
        __syncthreads();
    }
    if (tid == 0) esc[b * TIN + t] = red[0] + A2b[0];
}

// ---------------------------------------------------------------------------
// Kernel 4: softmax over t + pooled attended[b][j]
// ---------------------------------------------------------------------------
__global__ __launch_bounds__(512) void k_att_pool(
    const float* __restrict__ h2s, const float* __restrict__ esc,
    float* __restrict__ att)
{
    const int b = blockIdx.x, j = threadIdx.x;
    float e[TIN]; float m = -3.4028235e38f;
#pragma unroll
    for (int t = 0; t < TIN; ++t) { e[t] = esc[b * TIN + t]; m = fmaxf(m, e[t]); }
    float s = 0.f;
#pragma unroll
    for (int t = 0; t < TIN; ++t) { e[t] = __expf(e[t] - m); s += e[t]; }
    const float inv = 1.f / s;
    float acc = 0.f;
#pragma unroll
    for (int t = 0; t < TIN; ++t)
        acc += (e[t] * inv) * h2s[t * (BATCH * HID) + b * HID + j];
    att[b * HID + j] = acc;
}

// ---------------------------------------------------------------------------
// Kernel 5: feats[b][j] = attended[b] . Pdw[j] + Pdb[j]   (4 x 9000)
// grid 563 (16-col tiles), 1 wave per block
// ---------------------------------------------------------------------------
__global__ __launch_bounds__(32) void k_predec(
    const float* __restrict__ att, const float* __restrict__ Pdw,
    const float* __restrict__ Pdb, float* __restrict__ feats)
{
    __shared__ __attribute__((aligned(32))) _Float16 A[16 * 32];
    __shared__ __attribute__((aligned(32))) _Float16 Bt[16 * 32];
    const int lane = threadIdx.x;
    const int j0 = blockIdx.x * 16;
    v8f c = {};

    if (lane >= BATCH && lane < 16) {
        const v16h z = {};
        *(v16h*)(A + lane * 32)      = z;
        *(v16h*)(A + lane * 32 + 16) = z;
    }

    for (int ch = 0; ch < 16; ++ch) {
        const int k0 = ch * 32;
        if (lane < 8) {
            const int m = lane >> 1, hh = lane & 1;
            const float* src = att + m * HID + k0 + hh * 16;
            *(v16h*)(A + m * 32 + hh * 16) = pack16(src);
        }
        {
            const int n = lane & 15, hh = lane >> 4;
            const int j = j0 + n;
            if (j < FEAT) {
                const float* src = Pdw + (size_t)j * HID + k0 + hh * 16;
                *(v16h*)(Bt + n * 32 + hh * 16) = pack16(src);
            } else {
                const v16h z = {};
                *(v16h*)(Bt + n * 32 + hh * 16) = z;
            }
        }
        wave_lds_fence();
        v16h af = frag_A(A, 32);
        v16h bf = frag_B(Bt, 32);
        c = __builtin_amdgcn_wmma_f32_16x16x32_f16(false, af, false, bf, (short)0, c, false, false);
    }
    {
        const int grp = lane >> 4, n = lane & 15;
        const int j = j0 + n;
        if (grp == 0 && j < FEAT) {
#pragma unroll
            for (int r = 0; r < BATCH; ++r)
                feats[r * FEAT + j] = c[r] + Pdb[j];
        }
    }
}

// ---------------------------------------------------------------------------
// Kernel 6: per-batch stable descending argsort (bitonic, idx tie-break) +
// greedy sequential NMS + max-events truncation + masked output write.
// grid 4 (one per batch; masks identical across TOUT), 1024 thr
// ---------------------------------------------------------------------------
__global__ __launch_bounds__(1024) void k_nms(
    const float* __restrict__ feats, float* __restrict__ out)
{
    __shared__ float          key[NSORT];
    __shared__ unsigned short sidx[NSORT];
    __shared__ unsigned       kbits[NSORT / 32];
    __shared__ unsigned       vbits[(NVOX + 31) / 32];
    __shared__ int nexti;
    __shared__ int scount;
    const int b = blockIdx.x, tid = threadIdx.x;
    const float* fb = feats + b * FEAT;

    for (int p = tid; p < NSORT; p += 1024) {
        key[p]  = (p < NVOX) ? fb[p] : -3.4028235e38f;   // channel 0 = energy
        sidx[p] = (unsigned short)p;
    }
    __syncthreads();
    // bitonic: descending by key, ascending idx on ties (== stable argsort desc)
    for (int k = 2; k <= NSORT; k <<= 1) {
        for (int j = k >> 1; j > 0; j >>= 1) {
            for (int p = tid; p < NSORT; p += 1024) {
                const int q = p ^ j;
                if (q > p) {
                    const float k1 = key[p], k2 = key[q];
                    const unsigned short i1 = sidx[p], i2 = sidx[q];
                    const bool before = (k1 > k2) || (k1 == k2 && i1 < i2);
                    const bool want   = ((p & k) == 0);
                    if (before != want) {
                        key[p] = k2; key[q] = k1;
                        sidx[p] = i2; sidx[q] = i1;
                    }
                }
            }
            __syncthreads();
        }
    }
    // init keep bits (only sorted positions < NVOX)
    for (int wd = tid; wd < NSORT / 32; wd += 1024) {
        const int base = wd * 32;
        unsigned v = 0u;
        if (base + 32 <= NVOX) v = 0xFFFFFFFFu;
        else if (base < NVOX)  v = (1u << (NVOX - base)) - 1u;
        kbits[wd] = v;
    }
    for (int wd = tid; wd < (NVOX + 31) / 32; wd += 1024) vbits[wd] = 0u;
    if (tid == 0) { nexti = 0; scount = 0; }
    __syncthreads();

    // greedy NMS: iterate kept positions only
    int cur = -1;
    for (;;) {
        if (tid == 0) nexti = 0x7FFFFFFF;
        __syncthreads();
        const int start = cur + 1;
        if (tid < NSORT / 32) {
            const int base = tid * 32;
            if (base + 31 >= start) {
                unsigned bits = kbits[tid];
                if (base < start) bits &= ~((1u << (start - base)) - 1u);
                if (bits) atomicMin(&nexti, base + __ffs(bits) - 1);
            }
        }
        __syncthreads();
        const int i = nexti;
        if (i >= NVOX) break;
        const int idi = sidx[i];
        const int xi = idi / (D2 * D3), yi = (idi % (D2 * D3)) / D3, zi = idi % D3;
        for (int p = i + 1 + tid; p < NVOX; p += 1024) {
            if (kbits[p >> 5] & (1u << (p & 31))) {
                const int idj = sidx[p];
                const int dx = xi - idj / (D2 * D3);
                const int dy = yi - (idj % (D2 * D3)) / D3;
                const int dz = zi - idj % D3;
                if (dx * dx + dy * dy + dz * dz < 4)   // dist < 2.0
                    atomicAnd(&kbits[p >> 5], ~(1u << (p & 31)));
            }
        }
        __syncthreads();
        cur = i;
    }
    // truncate to first 100 sorted positions iff count > 100
    if (tid < NSORT / 32) atomicAdd(&scount, __popc(kbits[tid]));
    __syncthreads();
    if (scount > 100) {
        if (tid < NSORT / 32) {
            if (tid == 3)      kbits[3] &= 0xFu;   // keep positions 96..99
            else if (tid > 3)  kbits[tid] = 0u;
        }
    }
    __syncthreads();
    // scatter sorted-position keeps back to voxel mask
    for (int p = tid; p < NVOX; p += 1024) {
        if (kbits[p >> 5] & (1u << (p & 31))) {
            const int id = sidx[p];
            atomicOr(&vbits[id >> 5], 1u << (id & 31));
        }
    }
    __syncthreads();
    // write output (b, t=0..2, ch, v); every element written every call
    for (int e = tid; e < TOUT * 2 * NVOX; e += 1024) {
        const int t = e / (2 * NVOX), r = e % (2 * NVOX);
        const int ch = r / NVOX, v = r % NVOX;
        const float val = (vbits[v >> 5] & (1u << (v & 31)))
                            ? fb[ch * NVOX + v] : 0.f;
        out[((size_t)(b * TOUT + t)) * FEAT + ch * NVOX + v] = val;
    }
}

// ---------------------------------------------------------------------------
extern "C" void kernel_launch(void* const* d_in, const int* in_sizes, int n_in,
                              void* d_out, int out_size, void* d_ws, size_t ws_size,
                              hipStream_t stream) {
    (void)in_sizes; (void)n_in; (void)out_size; (void)ws_size;
    const float* x    = (const float*)d_in[0];
    const float* Wih0 = (const float*)d_in[1];
    const float* Whh0 = (const float*)d_in[2];
    const float* bih0 = (const float*)d_in[3];
    const float* bhh0 = (const float*)d_in[4];
    const float* Wih1 = (const float*)d_in[5];
    const float* Whh1 = (const float*)d_in[6];
    const float* bih1 = (const float*)d_in[7];
    const float* bhh1 = (const float*)d_in[8];
    const float* A1w  = (const float*)d_in[9];
    const float* A1b  = (const float*)d_in[10];
    const float* A2w  = (const float*)d_in[11];
    const float* A2b  = (const float*)d_in[12];
    const float* Pdw  = (const float*)d_in[13];
    const float* Pdb  = (const float*)d_in[14];

    float* ws   = (float*)d_ws;
    float* G0   = ws;                  // 20*2048            = 40960
    float* h1s  = ws + 40960;          // 5*4*512            = 10240
    float* h2s  = ws + 51200;          // 5*4*512            = 10240
    float* c0   = ws + 61440;          // 4*512              =  2048
    float* c1   = ws + 63488;          // 4*512              =  2048
    float* esc  = ws + 65536;          // 20 (+pad)          =    32
    float* att  = ws + 65568;          // 4*512              =  2048
    float* feats= ws + 67616;          // 4*2*4500           = 36000
    float* out  = (float*)d_out;

    k_gemm_g0<<<128, 256, 0, stream>>>(x, Wih0, bih0, bhh0, G0);
    for (int t = 0; t < TIN; ++t) {
        // layer 0: x-side precomputed in G0 (includes both biases)
        k_lstm_step<<<32, 128, 0, stream>>>(
            nullptr, nullptr, G0 + t * (BATCH * GATES),
            (t ? h1s + (t - 1) * (BATCH * HID) : nullptr), Whh0,
            nullptr, nullptr, c0, h1s + t * (BATCH * HID), t == 0);
        // layer 1: x-side = h1[t] @ Wih1^T computed in-kernel
        k_lstm_step<<<32, 128, 0, stream>>>(
            h1s + t * (BATCH * HID), Wih1, nullptr,
            (t ? h2s + (t - 1) * (BATCH * HID) : nullptr), Whh1,
            bih1, bhh1, c1, h2s + t * (BATCH * HID), t == 0);
    }
    k_att_scores<<<20, 256, 0, stream>>>(h2s, A1w, A1b, A2w, A2b, esc);
    k_att_pool<<<BATCH, 512, 0, stream>>>(h2s, esc, att);
    k_predec<<<(FEAT + 15) / 16, 32, 0, stream>>>(att, Pdw, Pdb, feats);
    k_nms<<<BATCH, 1024, 0, stream>>>(feats, out);
}
